// EinsteinCore_13099650252823
// MI455X (gfx1250) — matrix operand order repaired and checked by reading, not compiled, not run on hardware
//
#include <hip/hip_runtime.h>

// ---------------------------------------------------------------------------
// Hyperbolic (Lorentz) attention for MI455X / gfx1250, wave32 + WMMA bf16.
// B=8, S=M=2048, STATE=MEM=1024, DH=DOUT=512, k=1 (sk=1), BETA=1/sqrt(512).
//
// Pipeline (all on `stream`):
//   wconv   : f32 -> bf16 weight conversion (Wq, Wk, Wv)
//   proj    : X @ W^T + b, expmap0 -> spatial xs (bf16) + x0 (f32)
//             (for V: gamma*v_klein == xs, gamma == x0 exactly; xs stored
//              transposed (B, DOUT, M) so aggregation B-fragments are
//              contiguous per-lane 32B loads)
//   attn    : flash-style fused scores -> arccosh -> online softmax ->
//             Einstein-midpoint aggregation -> logmap0 epilogue
// ---------------------------------------------------------------------------

typedef __attribute__((ext_vector_type(16))) __bf16 v16bf;
typedef __attribute__((ext_vector_type(8)))  float  v8f;

union FragBF { v16bf v; uint4 u[2]; };

#if __has_builtin(__builtin_amdgcn_global_load_async_to_lds_b128)
#define EC_HAS_ASYNC_LDS 1
// builtin expects pointers to 16-byte int vectors in AS1 (global) / AS3 (LDS)
typedef int ec_v4i __attribute__((vector_size(16)));
typedef __attribute__((address_space(1))) ec_v4i* ec_gptr;
typedef __attribute__((address_space(3))) ec_v4i* ec_lptr;
#else
#define EC_HAS_ASYNC_LDS 0
#endif

__device__ __forceinline__ unsigned short f2bf(float f) {
  unsigned int x = __builtin_bit_cast(unsigned int, f);
  unsigned int r = x + 0x7fffu + ((x >> 16) & 1u);   // round-to-nearest-even
  return (unsigned short)(r >> 16);
}
__device__ __forceinline__ unsigned int pack2(float a, float b) {
  return (unsigned int)f2bf(a) | ((unsigned int)f2bf(b) << 16);
}

// cross-lane xor shuffle via ds_swizzle (group-of-32 mode, and=0x1f, xor=MASK)
template <int MASK>
__device__ __forceinline__ float swz(float x) {
  int i = __builtin_bit_cast(int, x);
  i = __builtin_amdgcn_ds_swizzle(i, (MASK << 10) | 0x1f);
  return __builtin_bit_cast(float, i);
}
__device__ __forceinline__ float redmax16(float v) {   // reduce lanes 0-15 / 16-31 independently
  v = fmaxf(v, swz<1>(v)); v = fmaxf(v, swz<2>(v));
  v = fmaxf(v, swz<4>(v)); v = fmaxf(v, swz<8>(v));
  return v;
}
__device__ __forceinline__ float redsum16(float v) {
  v += swz<1>(v); v += swz<2>(v); v += swz<4>(v); v += swz<8>(v);
  return v;
}

#define WMMA_BF16(A, Bm, C) \
  __builtin_amdgcn_wmma_f32_16x16x32_bf16(false, (A), false, (Bm), (short)0, (C), false, false)

// ---------------------------------------------------------------------------
// Kernel 0: f32 -> bf16 elementwise (weights)
// ---------------------------------------------------------------------------
__global__ __launch_bounds__(256) void EC_wconv(const float4* __restrict__ in,
                                                uint2* __restrict__ out, int n4) {
  int i = blockIdx.x * 256 + threadIdx.x;
  if (i < n4) {
    float4 v = in[i];
    out[i] = make_uint2(pack2(v.x, v.y), pack2(v.z, v.w));
  }
}

// ---------------------------------------------------------------------------
// Kernel 1: projection + expmap0.
//   X  : (B*R, 1024) f32 row tile of 16 rows per workgroup
//   Wb : (512, 1024) bf16 row-major (so B fragments are contiguous)
//   Xs : spatial output bf16; transposed==0 -> (B*R, 512)
//                            transposed==1 -> (B, 512, 2048)  (DOUT-major)
//   x0 : (B*R) f32 time component (== gamma for V)
// 8 waves; wave w owns 64-wide output slice (4 n-tiles), K loop over 1024.
// ---------------------------------------------------------------------------
__global__ __launch_bounds__(256) void EC_proj(const float* __restrict__ X,
                                               const unsigned short* __restrict__ Wb,
                                               const float* __restrict__ bias,
                                               unsigned short* __restrict__ Xs,
                                               float* __restrict__ x0,
                                               int transposed) {
  __shared__ __align__(16) unsigned short sX[16 * 1024];
  __shared__ float sNorm[16 * 8];
  __shared__ float sCoef[16];

  const int tid  = threadIdx.x;
  const int wave = tid >> 5, lane = tid & 31;
  const int lrow = lane & 15, hi = lane >> 4;
  const int rowbase = blockIdx.x * 16;     // global row (flattened over B)

  // stage input tile as bf16 into LDS (flat 16x1024)
  #pragma unroll
  for (int i = 0; i < 16; ++i) {
    int f = (tid + i * 256) * 4;
    float4 v = *(const float4*)&X[(size_t)rowbase * 1024 + f];
    *(uint2*)&sX[f] = make_uint2(pack2(v.x, v.y), pack2(v.z, v.w));
  }
  __syncthreads();

  const int nbase = wave * 64;
  v8f a[4] = {};

  for (int kb = 0; kb < 1024; kb += 32) {
    FragBF fa;
    int ai = lrow * 1024 + kb + hi * 16;           // A: lane<16 K0..15
    fa.u[0] = *(const uint4*)&sX[ai];
    fa.u[1] = *(const uint4*)&sX[ai + 8];
    #pragma unroll
    for (int nt = 0; nt < 4; ++nt) {
      FragBF fb;
      size_t bi = (size_t)(nbase + nt * 16 + lrow) * 1024 + kb + hi * 16;
      fb.u[0] = *(const uint4*)&Wb[bi];            // B: lane<16 K0..15 of col
      fb.u[1] = *(const uint4*)&Wb[bi + 8];
      a[nt] = WMMA_BF16(fa.v, fb.v, a[nt]);
    }
  }

  // bias + per-row squared norms (this wave's 64 columns)
  float bv[4];
  #pragma unroll
  for (int nt = 0; nt < 4; ++nt) bv[nt] = bias[nbase + nt * 16 + lrow];
  float ss[8];
  #pragma unroll
  for (int j = 0; j < 8; ++j) {
    float s = 0.f;
    #pragma unroll
    for (int nt = 0; nt < 4; ++nt) { a[nt][j] += bv[nt]; s += a[nt][j] * a[nt][j]; }
    ss[j] = redsum16(s);
  }
  if (lrow == 0) {
    #pragma unroll
    for (int j = 0; j < 8; ++j) sNorm[(j + hi * 8) * 8 + wave] = ss[j];
  }
  __syncthreads();

  if (tid < 16) {
    float n2 = 0.f;
    #pragma unroll
    for (int w = 0; w < 8; ++w) n2 += sNorm[tid * 8 + w];
    float n    = __builtin_sqrtf(n2);
    float smin = fminf(3.5f / (n + 1e-7f), 1.0f);  // tangent-norm clamp
    float vn   = fmaxf(n * smin, 1e-12f);          // theta (sk==1)
    x0[rowbase + tid] = coshf(vn);                 // time component / gamma
    sCoef[tid] = sinhf(vn) / vn * smin;            // xs = coef * x
  }
  __syncthreads();

  #pragma unroll
  for (int j = 0; j < 8; ++j) {
    float cf = sCoef[j + hi * 8];
    int g = rowbase + j + hi * 8;
    #pragma unroll
    for (int nt = 0; nt < 4; ++nt) {
      unsigned short o = f2bf(a[nt][j] * cf);
      int d = nbase + nt * 16 + lrow;
      if (!transposed)
        Xs[(size_t)g * 512 + d] = o;
      else                                           // (B, DOUT, M)
        Xs[((size_t)((g >> 11) * 512 + d)) * 2048 + (g & 2047)] = o;
    }
  }
}

// ---------------------------------------------------------------------------
// Kernel 2: fused flash-style hyperbolic attention.
// One workgroup (8 waves) = 16 query rows. Per 128-wide M tile:
//   wave w -> 16x16 score tile of columns [mb+16w, mb+16w+16)
//   online softmax (cross-lane ds_swizzle + cross-wave LDS)
//   wave w -> accumulates 64-wide DOUT slice of P @ W^T
// ---------------------------------------------------------------------------
__global__ __launch_bounds__(256) void EC_attn(const unsigned short* __restrict__ Qs,
                                               const float* __restrict__ q0,
                                               const unsigned short* __restrict__ Ks,
                                               const float* __restrict__ k0,
                                               const unsigned short* __restrict__ WT,
                                               const float* __restrict__ g0,
                                               float* __restrict__ out) {
  __shared__ __align__(16) unsigned short sQ[16 * 512];
  __shared__ __align__(16) unsigned short sP[16 * 128];
  __shared__ float sRowMax[16 * 8];
  __shared__ float sDenomPart[16 * 8];
  __shared__ float sNormPart[16 * 8];
  __shared__ float sR[16], sNewMax[16], sRunMax[16], sRunDenom[16], sQ0[16], sCoef[16];

  const int tid  = threadIdx.x;
  const int wave = tid >> 5, lane = tid & 31;
  const int lrow = lane & 15, hi = lane >> 4;
  const int tileRow = blockIdx.x * 16;      // global (b*S + s) row base
  const int b = tileRow >> 11;              // S == 2048

  // stage Q tile (16x512 bf16): async memory->LDS (no VGPR round-trip)
#if EC_HAS_ASYNC_LDS
  #pragma unroll
  for (int i = 0; i < 4; ++i) {
    int idx = (tid + i * 256) * 8;
    __builtin_amdgcn_global_load_async_to_lds_b128(
        (ec_gptr)(Qs + (size_t)tileRow * 512 + idx),
        (ec_lptr)&sQ[idx], 0, 0);
  }
#else
  #pragma unroll
  for (int i = 0; i < 4; ++i) {
    int idx = (tid + i * 256) * 8;
    *(uint4*)&sQ[idx] = *(const uint4*)&Qs[(size_t)tileRow * 512 + idx];
  }
#endif
  if (tid < 16) {
    sQ0[tid] = q0[tileRow + tid];
    sRunMax[tid] = -3.0e38f;
    sRunDenom[tid] = 0.f;
  }
  v8f acc[4] = {};
#if EC_HAS_ASYNC_LDS
#if __has_builtin(__builtin_amdgcn_s_wait_asynccnt)
  __builtin_amdgcn_s_wait_asynccnt(0);
#else
  asm volatile("s_wait_asynccnt 0" ::: "memory");
#endif
#endif
  __syncthreads();

  for (int mb = 0; mb < 2048; mb += 128) {
    const int mc = mb + wave * 16;

    // prefetch next M tile's K rows / W^T rows toward the WGP while we compute
    if (mb + 128 < 2048) {
      __builtin_prefetch(&Ks[(size_t)(b * 2048 + mc + 128 + lrow) * 512 + hi * 256], 0, 3);
      __builtin_prefetch(&WT[(size_t)(b * 512 + wave * 64 + lrow) * 2048 + mb + 128 + hi * 64], 0, 3);
    }

    // ---- scores: inner = Qs . Ks over DH=512 (16 bf16 WMMAs)
    v8f c = {};
    for (int kb = 0; kb < 512; kb += 32) {
      FragBF fa, fb;
      int ai = lrow * 512 + kb + hi * 16;
      fa.u[0] = *(const uint4*)&sQ[ai];
      fa.u[1] = *(const uint4*)&sQ[ai + 8];
      size_t bi = (size_t)(b * 2048 + mc + lrow) * 512 + kb + hi * 16;
      fb.u[0] = *(const uint4*)&Ks[bi];
      fb.u[1] = *(const uint4*)&Ks[bi + 8];
      c = WMMA_BF16(fa.v, fb.v, c);
    }

    // logits = -BETA * arccosh(max(-(inner - q0*k0), 1+1e-7))
    const float k0v = k0[b * 2048 + mc + lrow];
    float lg[8], lmax[8];
    #pragma unroll
    for (int j = 0; j < 8; ++j) {
      float inner = c[j] - sQ0[j + hi * 8] * k0v;
      float x = fmaxf(-inner, 1.0f + 1e-7f);
      float d = __logf(x + __builtin_sqrtf(x * x - 1.0f));
      lg[j]   = -0.044194173824159216f * d;        // BETA = 1/sqrt(512)
      lmax[j] = redmax16(lg[j]);
    }
    if (lrow == 0) {
      #pragma unroll
      for (int j = 0; j < 8; ++j) sRowMax[(j + hi * 8) * 8 + wave] = lmax[j];
    }
    __syncthreads();

    if (tid < 16) {                                 // online max update
      float tm = sRowMax[tid * 8];
      #pragma unroll
      for (int w = 1; w < 8; ++w) tm = fmaxf(tm, sRowMax[tid * 8 + w]);
      float nm = fmaxf(sRunMax[tid], tm);
      sR[tid] = __expf(sRunMax[tid] - nm);
      sNewMax[tid] = nm;
      sRunMax[tid] = nm;
    }
    __syncthreads();

    // p = exp(l - newmax); P tile to LDS (bf16); denom partial = sum p*gamma
    const float gv = g0[b * 2048 + mc + lrow];
    float dsum[8];
    #pragma unroll
    for (int j = 0; j < 8; ++j) {
      float p = __expf(lg[j] - sNewMax[j + hi * 8]);
      sP[(j + hi * 8) * 128 + wave * 16 + lrow] = f2bf(p);
      dsum[j] = redsum16(p * gv);
    }
    if (lrow == 0) {
      #pragma unroll
      for (int j = 0; j < 8; ++j) sDenomPart[(j + hi * 8) * 8 + wave] = dsum[j];
    }
    // rescale running accumulators by exp(oldmax - newmax)
    #pragma unroll
    for (int j = 0; j < 8; ++j) {
      float r = sR[j + hi * 8];
      #pragma unroll
      for (int nt = 0; nt < 4; ++nt) acc[nt][j] *= r;
    }
    __syncthreads();

    if (tid < 16) {                                 // online denom update
      float s = 0.f;
      #pragma unroll
      for (int w = 0; w < 8; ++w) s += sDenomPart[tid * 8 + w];
      sRunDenom[tid] = sRunDenom[tid] * sR[tid] + s;
    }

    // ---- aggregation: acc += P(16x128) @ W^T slice(128x64)  (16 WMMAs)
    for (int ks = 0; ks < 4; ++ks) {
      FragBF fa;
      int ai = lrow * 128 + ks * 32 + hi * 16;
      fa.u[0] = *(const uint4*)&sP[ai];
      fa.u[1] = *(const uint4*)&sP[ai + 8];
      #pragma unroll
      for (int nt = 0; nt < 4; ++nt) {
        FragBF fb;
        size_t drow = (size_t)(b * 512 + wave * 64 + nt * 16 + lrow);
        size_t bi = drow * 2048 + mb + ks * 32 + hi * 16;
        fb.u[0] = *(const uint4*)&WT[bi];
        fb.u[1] = *(const uint4*)&WT[bi + 8];
        acc[nt] = WMMA_BF16(fa.v, fb.v, acc[nt]);
      }
    }
    __syncthreads();    // protect sP / sDenomPart / sRowMax for next M tile
  }

  // ---- epilogue: mid = acc/denom; out = (arccosh(z0)/||mid||) * mid
  float ss[8];
  #pragma unroll
  for (int j = 0; j < 8; ++j) {
    float dn = fmaxf(sRunDenom[j + hi * 8], 1e-12f);
    float s = 0.f;
    #pragma unroll
    for (int nt = 0; nt < 4; ++nt) { acc[nt][j] /= dn; s += acc[nt][j] * acc[nt][j]; }
    ss[j] = redsum16(s);
  }
  if (lrow == 0) {
    #pragma unroll
    for (int j = 0; j < 8; ++j) sNormPart[(j + hi * 8) * 8 + wave] = ss[j];
  }
  __syncthreads();
  if (tid < 16) {
    float n2 = 0.f;
    #pragma unroll
    for (int w = 0; w < 8; ++w) n2 += sNormPart[tid * 8 + w];
    float n   = __builtin_sqrtf(n2);
    float s1  = (n >= 1.0f) ? (1.0f - 1e-6f) / n : 1.0f;     // Klein ball clamp
    float mn2 = fminf(n2 * s1 * s1, 1.0f - 1e-6f);
    float z0  = 1.0f / __builtin_sqrtf(fmaxf(1.0f - mn2, 1e-12f));
    float x   = fmaxf(z0, 1.0f + 1e-7f);
    float d0  = __logf(x + __builtin_sqrtf(x * x - 1.0f));   // arccosh
    sCoef[tid] = d0 / fmaxf(n, 1e-12f);
  }
  __syncthreads();
  #pragma unroll
  for (int j = 0; j < 8; ++j) {
    float cc = sCoef[j + hi * 8];
    size_t g = (size_t)(tileRow + j + hi * 8);
    #pragma unroll
    for (int nt = 0; nt < 4; ++nt)
      out[g * 512 + wave * 64 + nt * 16 + lrow] = acc[nt][j] * cc;
  }
}

// ---------------------------------------------------------------------------
// Host-side launch. Workspace layout (bytes):
//   [0,1M)   WqB bf16   [1M,2M) WkB   [2M,3M) WvB
//   [4M,20M)  Qs bf16 (B*S,512)
//   [20M,36M) Ks bf16 (B*M,512)
//   [36M,52M) WT bf16 (B,512,2048)   (gamma*v_klein, transposed)
//   [52M,..)  q0 | k0 | g0  (64KB each, f32)
// ---------------------------------------------------------------------------
extern "C" void kernel_launch(void* const* d_in, const int* in_sizes, int n_in,
                              void* d_out, int out_size, void* d_ws, size_t ws_size,
                              hipStream_t stream) {
  (void)in_sizes; (void)n_in; (void)out_size; (void)ws_size;

  const float* queries = (const float*)d_in[0];
  const float* keys    = (const float*)d_in[1];
  const float* values  = (const float*)d_in[2];
  const float* Wq      = (const float*)d_in[3];
  const float* bq      = (const float*)d_in[4];
  const float* Wk      = (const float*)d_in[5];
  const float* bk      = (const float*)d_in[6];
  const float* Wv      = (const float*)d_in[7];
  const float* bv      = (const float*)d_in[8];

  char* ws = (char*)d_ws;
  const size_t MB = 1u << 20;
  unsigned short* WqB = (unsigned short*)(ws + 0 * MB);
  unsigned short* WkB = (unsigned short*)(ws + 1 * MB);
  unsigned short* WvB = (unsigned short*)(ws + 2 * MB);
  unsigned short* Qs  = (unsigned short*)(ws + 4 * MB);
  unsigned short* Ks  = (unsigned short*)(ws + 20 * MB);
  unsigned short* WT  = (unsigned short*)(ws + 36 * MB);
  float* q0 = (float*)(ws + 52 * MB);
  float* k0 = (float*)(ws + 52 * MB + 64 * 1024);
  float* g0 = (float*)(ws + 52 * MB + 128 * 1024);

  const int n4 = 512 * 1024 / 4;
  EC_wconv<<<dim3(n4 / 256), dim3(256), 0, stream>>>((const float4*)Wq, (uint2*)WqB, n4);
  EC_wconv<<<dim3(n4 / 256), dim3(256), 0, stream>>>((const float4*)Wk, (uint2*)WkB, n4);
  EC_wconv<<<dim3(n4 / 256), dim3(256), 0, stream>>>((const float4*)Wv, (uint2*)WvB, n4);

  // 8*2048/16 = 1024 row tiles per projection
  EC_proj<<<dim3(1024), dim3(256), 0, stream>>>(queries, WqB, bq, Qs, q0, 0);
  EC_proj<<<dim3(1024), dim3(256), 0, stream>>>(keys,    WkB, bk, Ks, k0, 0);
  EC_proj<<<dim3(1024), dim3(256), 0, stream>>>(values,  WvB, bv, WT, g0, 1);

  EC_attn<<<dim3(1024), dim3(256), 0, stream>>>(Qs, q0, Ks, k0, WT, g0, (float*)d_out);
}
